// Discriminator_42039139894156
// MI455X (gfx1250) — compile-verified, matrix operand order checked by reading, not run on hardware
//
#include <hip/hip_runtime.h>
#include <hip/hip_fp16.h>

#define WDIM  28
#define HDIM  28
#define BATCH 128
#define NPTS  512
#define EPSV  1e-5f

typedef __attribute__((ext_vector_type(16))) _Float16 v16h;
typedef __attribute__((ext_vector_type(8)))  float    v8f;

// ---------------------------------------------------------------------------
// 1) Wireframe: scatter-max of <=2x2 footprints into an LDS image per batch.
//    All contributions are >= 0, so float bits are order-preserving and we
//    can use u32 LDS atomic max (ds_max_u32).
// ---------------------------------------------------------------------------
__global__ __launch_bounds__(256) void wireframe_kernel(const float* __restrict__ x_in,
                                                        float* __restrict__ out) {
  __shared__ unsigned img[WDIM * HDIM];
  const int b = blockIdx.x;
  for (int i = threadIdx.x; i < WDIM * HDIM; i += blockDim.x) img[i] = 0u;
  __syncthreads();
  for (int e = threadIdx.x; e < NPTS; e += blockDim.x) {
    const float* q = x_in + ((size_t)b * NPTS + e) * 3;
    float p  = q[0];
    float tx = q[1] * (float)WDIM;
    float ty = q[2] * (float)HDIM;
    int cx = (int)floorf(tx);
    int cy = (int)floorf(ty);
#pragma unroll
    for (int dx = 0; dx < 2; ++dx) {
      int gx = cx + dx;
      if (gx < 0 || gx >= WDIM) continue;
      float kx = 1.0f - fabsf((float)gx - tx);
      if (kx <= 0.0f) continue;
#pragma unroll
      for (int dy = 0; dy < 2; ++dy) {
        int gy = cy + dy;
        if (gy < 0 || gy >= HDIM) continue;
        float ky = 1.0f - fabsf((float)gy - ty);
        if (ky <= 0.0f) continue;
        float v = p * kx * ky;  // >= 0
        atomicMax(&img[gx * HDIM + gy], __float_as_uint(v));
      }
    }
  }
  __syncthreads();
  for (int i = threadIdx.x; i < WDIM * HDIM; i += blockDim.x)
    out[(size_t)b * WDIM * HDIM + i] = __uint_as_float(img[i]);
}

// ---------------------------------------------------------------------------
// 2) Generic small 3x3 SAME conv; one block per batch, input slab + weights
//    staged in LDS (max slab 784 floats, max weights 16*8*9=1152 floats).
// ---------------------------------------------------------------------------
__global__ __launch_bounds__(256) void conv3x3_kernel(const float* __restrict__ in,
                                                      const float* __restrict__ w,
                                                      const float* __restrict__ bias,
                                                      float* __restrict__ out,
                                                      int Cin, int Cout, int H, int W) {
  __shared__ float s_in[784];
  __shared__ float s_w[1152];
  __shared__ float s_b[16];
  const int b   = blockIdx.x;
  const int inN = Cin * H * W;
  const int wN  = Cout * Cin * 9;
  for (int i = threadIdx.x; i < inN; i += blockDim.x) s_in[i] = in[(size_t)b * inN + i];
  for (int i = threadIdx.x; i < wN;  i += blockDim.x) s_w[i]  = w[i];
  if ((int)threadIdx.x < Cout) s_b[threadIdx.x] = bias[threadIdx.x];
  __syncthreads();
  const int total = Cout * H * W;
  for (int idx = threadIdx.x; idx < total; idx += blockDim.x) {
    int co = idx / (H * W);
    int r  = idx % (H * W);
    int y  = r / W, x = r % W;
    float acc = s_b[co];
    for (int ci = 0; ci < Cin; ++ci) {
      const float* sp = s_in + ci * H * W;
      const float* wp = s_w + (co * Cin + ci) * 9;
#pragma unroll
      for (int ky = 0; ky < 3; ++ky) {
        int yy = y + ky - 1;
        if (yy < 0 || yy >= H) continue;
#pragma unroll
        for (int kx = 0; kx < 3; ++kx) {
          int xx = x + kx - 1;
          if (xx < 0 || xx >= W) continue;
          acc += sp[yy * W + xx] * wp[ky * 3 + kx];
        }
      }
    }
    out[(size_t)b * total + idx] = acc;
  }
}

// ---------------------------------------------------------------------------
// 3) BN train-mode statistics: one block per channel; writes (mean, invstd).
// ---------------------------------------------------------------------------
__global__ __launch_bounds__(256) void bn_stats_kernel(const float* __restrict__ x,
                                                       float* __restrict__ stats,
                                                       int C, int HW) {
  __shared__ float s0[256];
  __shared__ float s1[256];
  const int c = blockIdx.x;
  const int n = BATCH * HW;
  float sum = 0.f, sq = 0.f;
  for (int i = threadIdx.x; i < n; i += blockDim.x) {
    int b = i / HW, r = i % HW;
    float v = x[((size_t)b * C + c) * HW + r];
    sum += v;
    sq  += v * v;
  }
  s0[threadIdx.x] = sum;
  s1[threadIdx.x] = sq;
  __syncthreads();
  for (int s = 128; s > 0; s >>= 1) {
    if ((int)threadIdx.x < s) {
      s0[threadIdx.x] += s0[threadIdx.x + s];
      s1[threadIdx.x] += s1[threadIdx.x + s];
    }
    __syncthreads();
  }
  if (threadIdx.x == 0) {
    float mean = s0[0] / (float)n;
    float var  = s1[0] / (float)n - mean * mean;
    stats[2 * c]     = mean;
    stats[2 * c + 1] = rsqrtf(var + EPSV);
  }
}

// ---------------------------------------------------------------------------
// 4) Fused BN-affine + ReLU + (optional 2x2) max-pool epilogue.
// ---------------------------------------------------------------------------
__global__ __launch_bounds__(256) void bn_relu_pool_kernel(const float* __restrict__ x,
                                                           const float* __restrict__ stats,
                                                           const float* __restrict__ g,
                                                           const float* __restrict__ bt,
                                                           float* __restrict__ out,
                                                           int C, int H, int W, int pool) {
  const int b  = blockIdx.x;
  const int Ho = H / pool, Wo = W / pool;
  const int total = C * Ho * Wo;
  for (int idx = threadIdx.x; idx < total; idx += blockDim.x) {
    int co = idx / (Ho * Wo);
    int r  = idx % (Ho * Wo);
    int oy = r / Wo, ox = r % Wo;
    float mean = stats[2 * co], inv = stats[2 * co + 1];
    float sc = g[co] * inv;
    float sh = bt[co] - mean * sc;
    float best = -1e30f;
    for (int dy = 0; dy < pool; ++dy)
      for (int dx = 0; dx < pool; ++dx) {
        float v = x[((size_t)b * C + co) * H * W + (oy * pool + dy) * W + (ox * pool + dx)];
        v = fmaxf(v * sc + sh, 0.0f);
        best = fmaxf(best, v);
      }
    out[(size_t)b * total + idx] = best;
  }
}

// ---------------------------------------------------------------------------
// 5) Pack fc1 activations (128x784) and fc1 weights (128x784) to f16,
//    K padded 784 -> 800 (multiple of 32) with zeros.
// ---------------------------------------------------------------------------
__global__ __launch_bounds__(256) void pack_f16_kernel(const float* __restrict__ act,
                                                       const float* __restrict__ w,
                                                       __half* __restrict__ actH,
                                                       __half* __restrict__ wH) {
  int idx = blockIdx.x * blockDim.x + threadIdx.x;
  if (idx >= 2 * 128 * 800) return;
  int which = idx / (128 * 800);
  int r = (idx / 800) % 128;
  int k = idx % 800;
  const float* src = which ? w : act;
  __half* dst = which ? wH : actH;
  float v = (k < 784) ? src[r * 784 + k] : 0.0f;
  dst[r * 800 + k] = __float2half(v);
}

// ---------------------------------------------------------------------------
// 6) fc1 GEMM on the matrix engine: D(128x128) = A(128x800) * W^T(800x128),
//    v_wmma_f32_16x16x32_f16, one wave per 16x16 tile, 25 K-steps,
//    fused bias + ReLU epilogue.
//    A frag (ISA 7.12.2, 16-bit A 16x32): lane L -> row m=L&15,
//      vgpr i (pair 2i,2i+1): k = (i>>2)*16 + (L>>4)*8 + 2*(i&3).
//    B frag mirrors A (lane holds B column n = weight row), same K pattern.
//    C/D: element r of v8f -> m = r + 8*(L>>4), n = L&15.
// ---------------------------------------------------------------------------
__global__ __launch_bounds__(32) void fc1_wmma_kernel(const __half* __restrict__ A,
                                                      const __half* __restrict__ Bw,
                                                      const float* __restrict__ b1,
                                                      float* __restrict__ out) {
  const int lane   = threadIdx.x & 31;
  const int row    = lane & 15;
  const int halfid = lane >> 4;
  const int tm = blockIdx.x * 16;
  const int tn = blockIdx.y * 16;
  const unsigned* a32 = (const unsigned*)(A  + (size_t)(tm + row) * 800);
  const unsigned* b32 = (const unsigned*)(Bw + (size_t)(tn + row) * 800);
  union { v16h v; unsigned u[8]; } af, bf;
  v8f c = {};
  for (int kc = 0; kc < 800; kc += 32) {
    const int base0 = (kc + halfid * 8) >> 1;  // u32 units; k pairs are even-aligned
    const int base1 = base0 + 8;               // (kc + 16 + halfid*8) >> 1
#pragma unroll
    for (int j = 0; j < 4; ++j) {
      af.u[j]     = a32[base0 + j];
      af.u[4 + j] = a32[base1 + j];
      bf.u[j]     = b32[base0 + j];
      bf.u[4 + j] = b32[base1 + j];
    }
    c = __builtin_amdgcn_wmma_f32_16x16x32_f16(
        /*neg_a=*/false, af.v, /*neg_b=*/false, bf.v,
        /*c_mod=*/(short)0, c, /*reuse_a=*/false, /*reuse_b=*/false);
  }
#pragma unroll
  for (int r = 0; r < 8; ++r) {
    int m = tm + r + 8 * halfid;
    int n = tn + row;
    float y = fmaxf(c[r] + b1[n], 0.0f);
    out[(size_t)m * 128 + n] = y;
  }
}

// ---------------------------------------------------------------------------
// 7) fc2 + batch mean: out = (1/B) * sum_{b,j} relu_fc1[b][j]*w2[j] + b2.
// ---------------------------------------------------------------------------
__global__ __launch_bounds__(256) void final_kernel(const float* __restrict__ y,
                                                    const float* __restrict__ w2,
                                                    const float* __restrict__ b2,
                                                    float* __restrict__ out) {
  __shared__ float s[256];
  float acc = 0.f;
  for (int i = threadIdx.x; i < 128 * 128; i += blockDim.x)
    acc += y[i] * w2[i & 127];
  s[threadIdx.x] = acc;
  __syncthreads();
  for (int st = 128; st > 0; st >>= 1) {
    if ((int)threadIdx.x < st) s[threadIdx.x] += s[threadIdx.x + st];
    __syncthreads();
  }
  if (threadIdx.x == 0) out[0] = s[0] * (1.0f / 128.0f) + b2[0];
}

// ---------------------------------------------------------------------------
extern "C" void kernel_launch(void* const* d_in, const int* in_sizes, int n_in,
                              void* d_out, int out_size, void* d_ws, size_t ws_size,
                              hipStream_t stream) {
  (void)in_sizes; (void)n_in; (void)out_size; (void)ws_size;
  const float* x_in = (const float*)d_in[0];
  const float* c1w  = (const float*)d_in[1];
  const float* c1b  = (const float*)d_in[2];
  const float* g1   = (const float*)d_in[3];
  const float* bt1  = (const float*)d_in[4];
  const float* c2w  = (const float*)d_in[5];
  const float* c2b  = (const float*)d_in[6];
  const float* g2   = (const float*)d_in[7];
  const float* bt2  = (const float*)d_in[8];
  const float* c3w  = (const float*)d_in[9];
  const float* c3b  = (const float*)d_in[10];
  const float* g3   = (const float*)d_in[11];
  const float* bt3  = (const float*)d_in[12];
  const float* f1w  = (const float*)d_in[13];
  const float* f1b  = (const float*)d_in[14];
  const float* f2w  = (const float*)d_in[15];
  const float* f2b  = (const float*)d_in[16];
  float* outp = (float*)d_out;

  // Carve workspace (~4.7 MB total), 256B-aligned slices.
  char* ws = (char*)d_ws;
  size_t off = 0;
  auto carve = [&](size_t bytes) -> void* {
    void* p = ws + off;
    off += (bytes + 255) & ~(size_t)255;
    return p;
  };
  float*  img   = (float*)carve((size_t)BATCH * 784 * 4);
  float*  conv1 = (float*)carve((size_t)BATCH * 4 * 784 * 4);
  float*  pool1 = (float*)carve((size_t)BATCH * 4 * 196 * 4);
  float*  conv2 = (float*)carve((size_t)BATCH * 8 * 196 * 4);
  float*  pool2 = (float*)carve((size_t)BATCH * 8 * 49 * 4);
  float*  conv3 = (float*)carve((size_t)BATCH * 16 * 49 * 4);
  float*  act3  = (float*)carve((size_t)BATCH * 784 * 4);
  float*  st1   = (float*)carve(64);
  float*  st2   = (float*)carve(64);
  float*  st3   = (float*)carve(128);
  __half* actH  = (__half*)carve((size_t)BATCH * 800 * 2);
  __half* wH    = (__half*)carve((size_t)128 * 800 * 2);
  float*  fc1o  = (float*)carve((size_t)128 * 128 * 4);

  wireframe_kernel<<<BATCH, 256, 0, stream>>>(x_in, img);

  conv3x3_kernel<<<BATCH, 256, 0, stream>>>(img, c1w, c1b, conv1, 1, 4, 28, 28);
  bn_stats_kernel<<<4, 256, 0, stream>>>(conv1, st1, 4, 784);
  bn_relu_pool_kernel<<<BATCH, 256, 0, stream>>>(conv1, st1, g1, bt1, pool1, 4, 28, 28, 2);

  conv3x3_kernel<<<BATCH, 256, 0, stream>>>(pool1, c2w, c2b, conv2, 4, 8, 14, 14);
  bn_stats_kernel<<<8, 256, 0, stream>>>(conv2, st2, 8, 196);
  bn_relu_pool_kernel<<<BATCH, 256, 0, stream>>>(conv2, st2, g2, bt2, pool2, 8, 14, 14, 2);

  conv3x3_kernel<<<BATCH, 256, 0, stream>>>(pool2, c3w, c3b, conv3, 8, 16, 7, 7);
  bn_stats_kernel<<<16, 256, 0, stream>>>(conv3, st3, 16, 49);
  bn_relu_pool_kernel<<<BATCH, 256, 0, stream>>>(conv3, st3, g3, bt3, act3, 16, 7, 7, 1);

  pack_f16_kernel<<<(2 * 128 * 800 + 255) / 256, 256, 0, stream>>>(act3, f1w, actH, wH);
  fc1_wmma_kernel<<<dim3(8, 8), 32, 0, stream>>>(actH, wH, f1b, fc1o);
  final_kernel<<<1, 256, 0, stream>>>(fc1o, f2w, f2b, outp);
}
